// GAT_22548578304736
// MI455X (gfx1250) — matrix operand order repaired and checked by reading, not compiled, not run on hardware
//
#include <hip/hip_runtime.h>
#include <hip/hip_bf16.h>
#include <math.h>

// ---- problem constants (match reference) ----
#define IN_C   512
#define HEADS  8
#define HID    32
#define HIDC   256   // HEADS*HID
#define OUT_C  40
#define NEG_SLOPE 0.2f

typedef float     v2f  __attribute__((ext_vector_type(2)));
typedef float     v8f  __attribute__((ext_vector_type(8)));
typedef _Float16  v16h __attribute__((ext_vector_type(16)));

#if defined(__has_builtin)
#  if __has_builtin(__builtin_amdgcn_wmma_f32_16x16x4_f32)
#    define USE_WMMA_F32 1
#  endif
#endif
#ifndef USE_WMMA_F32
#  define USE_WMMA_F32 0
#endif

// ---- monotonic float<->uint key mapping for atomic segment-max ----
__device__ __forceinline__ unsigned f2key(float f) {
  unsigned b = __float_as_uint(f);
  return (b & 0x80000000u) ? ~b : (b | 0x80000000u);
}
__device__ __forceinline__ float key2f(unsigned k) {
  unsigned b = (k & 0x80000000u) ? (k ^ 0x80000000u) : ~k;
  return __uint_as_float(b);
}
__device__ __forceinline__ float leaky(float x) {
  return x > 0.f ? x : NEG_SLOPE * x;
}
__device__ __forceinline__ void edge_nodes(const long long* __restrict__ ei, int E,
                                           int e, int& s, int& d) {
  if (e < E) { s = (int)ei[e]; d = (int)ei[(long long)E + e]; }
  else       { s = d = e - E; }
}

// ---------------------------------------------------------------------------
// zero init
// ---------------------------------------------------------------------------
__global__ void zero_f(float* __restrict__ p, long long n) {
  long long i = (long long)blockIdx.x * blockDim.x + threadIdx.x;
  if (i < n) p[i] = 0.f;
}

// ---------------------------------------------------------------------------
// GEMM1: H[N,256] = X[N,512] @ W[512,256]   (WMMA, 16x16 tile / wave)
// block = (32,8): 8 waves; wave w does N-tiles w and w+8. grid.x = N/16 tiles.
// ---------------------------------------------------------------------------
__global__ void __launch_bounds__(256)
gemm1_wmma(const float* __restrict__ X, const float* __restrict__ W,
           float* __restrict__ H, int Nn) {
  const int lane  = threadIdx.x;            // 0..31
  const int wave  = threadIdx.y;            // 0..7
  const int mbase = blockIdx.x * 16;
  const int r     = lane & 15;
  const int hi    = lane >> 4;

  const int rowClamped = min(mbase + r, Nn - 1);
  const float* xrow = X + (long long)rowClamped * IN_C;

  const int n0 = wave * 16;
  const int n1 = (wave + 8) * 16;

  v8f acc0 = {};
  v8f acc1 = {};

#if USE_WMMA_F32
  const int kb = hi * 2;                    // K sub-offset: 0 or 2
  for (int k = 0; k < IN_C; k += 4) {
    v2f a;
    a.x = xrow[k + kb];
    a.y = xrow[k + kb + 1];
    const float* wk0 = W + (long long)(k + kb) * HIDC;
    const float* wk1 = wk0 + HIDC;
    v2f b0, b1;
    b0.x = wk0[n0 + r];  b0.y = wk1[n0 + r];
    b1.x = wk0[n1 + r];  b1.y = wk1[n1 + r];
    acc0 = __builtin_amdgcn_wmma_f32_16x16x4_f32(false, a, false, b0, (short)0, acc0, false, false);
    acc1 = __builtin_amdgcn_wmma_f32_16x16x4_f32(false, a, false, b1, (short)0, acc1, false, false);
  }
#else
  const int hi8 = hi * 8;
  for (int k = 0; k < IN_C; k += 32) {
    v16h a, b0, b1;
#pragma unroll
    for (int v = 0; v < 8; ++v) {
      const int kA = 2 * v + hi * 16;
      a[2 * v]     = (_Float16)xrow[k + kA];
      a[2 * v + 1] = (_Float16)xrow[k + kA + 1];
      const int kB = 2 * v + hi * 16;
      const float* wk0 = W + (long long)(k + kB) * HIDC;
      const float* wk1 = wk0 + HIDC;
      b0[2 * v]     = (_Float16)wk0[n0 + r];
      b0[2 * v + 1] = (_Float16)wk1[n0 + r];
      b1[2 * v]     = (_Float16)wk0[n1 + r];
      b1[2 * v + 1] = (_Float16)wk1[n1 + r];
    }
    acc0 = __builtin_amdgcn_wmma_f32_16x16x32_f16(false, a, false, b0, (short)0, acc0, false, false);
    acc1 = __builtin_amdgcn_wmma_f32_16x16x32_f16(false, a, false, b1, (short)0, acc1, false, false);
  }
#endif

#pragma unroll
  for (int i = 0; i < 8; ++i) {
    const int m = mbase + hi * 8 + i;
    if (m < Nn) {
      H[(long long)m * HIDC + n0 + r] = acc0[i];
      H[(long long)m * HIDC + n1 + r] = acc1[i];
    }
  }
}

// ---------------------------------------------------------------------------
// GEMM2: H2[N,40] = Hin[N,256] @ W2[256,40]  (WMMA, col-masked to 40)
// grid = (Ntiles, 3), block = 32 (one wave per 16x16 tile)
// ---------------------------------------------------------------------------
__global__ void __launch_bounds__(32)
gemm2_wmma(const float* __restrict__ Hin, const float* __restrict__ W2,
           float* __restrict__ H2, int Nn) {
  const int lane  = threadIdx.x;
  const int r     = lane & 15;
  const int hi    = lane >> 4;
  const int mbase = blockIdx.x * 16;
  const int nb    = blockIdx.y * 16;
  const int col   = nb + r;
  const bool colOK = (col < OUT_C);
  const int colSafe = colOK ? col : 0;

  const int rowClamped = min(mbase + r, Nn - 1);
  const float* xrow = Hin + (long long)rowClamped * HIDC;

  v8f acc = {};

#if USE_WMMA_F32
  const int kb = hi * 2;
  for (int k = 0; k < HIDC; k += 4) {
    v2f a, b;
    a.x = xrow[k + kb];
    a.y = xrow[k + kb + 1];
    float bx = W2[(long long)(k + kb) * OUT_C + colSafe];
    float by = W2[(long long)(k + kb + 1) * OUT_C + colSafe];
    b.x = colOK ? bx : 0.f;
    b.y = colOK ? by : 0.f;
    acc = __builtin_amdgcn_wmma_f32_16x16x4_f32(false, a, false, b, (short)0, acc, false, false);
  }
#else
  const int hi8 = hi * 8;
  for (int k = 0; k < HIDC; k += 32) {
    v16h a, b;
#pragma unroll
    for (int v = 0; v < 8; ++v) {
      const int kA = 2 * v + hi * 16;
      a[2 * v]     = (_Float16)xrow[k + kA];
      a[2 * v + 1] = (_Float16)xrow[k + kA + 1];
      const int kB = 2 * v + hi * 16;
      float bx = W2[(long long)(k + kB) * OUT_C + colSafe];
      float by = W2[(long long)(k + kB + 1) * OUT_C + colSafe];
      b[2 * v]     = (_Float16)(colOK ? bx : 0.f);
      b[2 * v + 1] = (_Float16)(colOK ? by : 0.f);
    }
    acc = __builtin_amdgcn_wmma_f32_16x16x32_f16(false, a, false, b, (short)0, acc, false, false);
  }
#endif

#pragma unroll
  for (int i = 0; i < 8; ++i) {
    const int m = mbase + hi * 8 + i;
    if (m < Nn && colOK) H2[(long long)m * OUT_C + col] = acc[i];
  }
}

// ---------------------------------------------------------------------------
// layer-1 attention scalars: a_s[n,h] = <H[n,h,:], att_src[h,:]>, same for dst
// ---------------------------------------------------------------------------
__global__ void node_att1(const float* __restrict__ H,
                          const float* __restrict__ as, const float* __restrict__ ad,
                          float* __restrict__ As, float* __restrict__ Ad, int Nn) {
  int i = blockIdx.x * blockDim.x + threadIdx.x;
  if (i >= Nn * HEADS) return;
  const int node = i >> 3, h = i & 7;
  const float* hp = H + (long long)node * HIDC + h * HID;
  const float* sp = as + h * HID;
  const float* dp = ad + h * HID;
  float vs = 0.f, vd = 0.f;
#pragma unroll
  for (int c = 0; c < HID; ++c) { float x = hp[c]; vs += x * sp[c]; vd += x * dp[c]; }
  As[i] = vs; Ad[i] = vd;
}

__global__ void edge_max1(const long long* __restrict__ ei, int E, int EE,
                          const float* __restrict__ As, const float* __restrict__ Ad,
                          unsigned* __restrict__ Mkey) {
  int idx = blockIdx.x * blockDim.x + threadIdx.x;
  if (idx >= EE * HEADS) return;
  int e = idx >> 3, h = idx & 7, s, d;
  edge_nodes(ei, E, e, s, d);
  float el = leaky(As[s * HEADS + h] + Ad[d * HEADS + h]);
  atomicMax(&Mkey[d * HEADS + h], f2key(el));
}

__global__ void edge_sum1(const long long* __restrict__ ei, int E, int EE,
                          const float* __restrict__ As, const float* __restrict__ Ad,
                          const unsigned* __restrict__ Mkey,
                          float* __restrict__ Z, float* __restrict__ Eexp) {
  int idx = blockIdx.x * blockDim.x + threadIdx.x;
  if (idx >= EE * HEADS) return;
  int e = idx >> 3, h = idx & 7, s, d;
  edge_nodes(ei, E, e, s, d);
  float el = leaky(As[s * HEADS + h] + Ad[d * HEADS + h]);
  float w  = __expf(el - key2f(Mkey[d * HEADS + h]));
  Eexp[idx] = w;
  atomicAdd(&Z[d * HEADS + h], w);
}

__global__ void edge_aggr1(const long long* __restrict__ ei, int E, int EE,
                           const float* __restrict__ Eexp, const float* __restrict__ Z,
                           const float* __restrict__ H, float* __restrict__ Out) {
  int idx = blockIdx.x * blockDim.x + threadIdx.x;   // e*256 + c
  if (idx >= EE * HIDC) return;
  int e = idx >> 8, c = idx & 255, h = c >> 5, s, d;
  edge_nodes(ei, E, e, s, d);
  float alpha = Eexp[e * HEADS + h] / (Z[d * HEADS + h] + 1e-16f);
  atomicAdd(&Out[(long long)d * HIDC + c], alpha * H[(long long)s * HIDC + c]);
}

// in-place: X = elu(X + b1)
__global__ void elu_bias(float* __restrict__ X, const float* __restrict__ b, int total) {
  int i = blockIdx.x * blockDim.x + threadIdx.x;
  if (i >= total) return;
  float v = X[i] + b[i & (HIDC - 1)];
  X[i] = v > 0.f ? v : expm1f(v);
}

// ---------------------------------------------------------------------------
// layer-2 attention (1 head, 40 channels)
// ---------------------------------------------------------------------------
__global__ void node_att2(const float* __restrict__ H2,
                          const float* __restrict__ as, const float* __restrict__ ad,
                          float* __restrict__ As, float* __restrict__ Ad, int Nn) {
  int i = blockIdx.x * blockDim.x + threadIdx.x;
  if (i >= Nn) return;
  const float* hp = H2 + (long long)i * OUT_C;
  float vs = 0.f, vd = 0.f;
#pragma unroll
  for (int c = 0; c < OUT_C; ++c) { float x = hp[c]; vs += x * as[c]; vd += x * ad[c]; }
  As[i] = vs; Ad[i] = vd;
}

__global__ void edge_max2(const long long* __restrict__ ei, int E, int EE,
                          const float* __restrict__ As, const float* __restrict__ Ad,
                          unsigned* __restrict__ Mkey) {
  int e = blockIdx.x * blockDim.x + threadIdx.x;
  if (e >= EE) return;
  int s, d; edge_nodes(ei, E, e, s, d);
  atomicMax(&Mkey[d], f2key(leaky(As[s] + Ad[d])));
}

__global__ void edge_sum2(const long long* __restrict__ ei, int E, int EE,
                          const float* __restrict__ As, const float* __restrict__ Ad,
                          const unsigned* __restrict__ Mkey,
                          float* __restrict__ Z, float* __restrict__ Eexp) {
  int e = blockIdx.x * blockDim.x + threadIdx.x;
  if (e >= EE) return;
  int s, d; edge_nodes(ei, E, e, s, d);
  float w = __expf(leaky(As[s] + Ad[d]) - key2f(Mkey[d]));
  Eexp[e] = w;
  atomicAdd(&Z[d], w);
}

__global__ void edge_aggr2(const long long* __restrict__ ei, int E, int EE,
                           const float* __restrict__ Eexp, const float* __restrict__ Z,
                           const float* __restrict__ H2, float* __restrict__ Out) {
  int idx = blockIdx.x * blockDim.x + threadIdx.x;   // e*40 + c
  if (idx >= EE * OUT_C) return;
  int e = idx / OUT_C, c = idx - e * OUT_C, s, d;
  edge_nodes(ei, E, e, s, d);
  float alpha = Eexp[e] / (Z[d] + 1e-16f);
  atomicAdd(&Out[(long long)d * OUT_C + c], alpha * H2[(long long)s * OUT_C + c]);
}

// in-place log_softmax(row + b2)
__global__ void finalize_lsm(float* __restrict__ out, const float* __restrict__ b2, int Nn) {
  int i = blockIdx.x * blockDim.x + threadIdx.x;
  if (i >= Nn) return;
  float* row = out + (long long)i * OUT_C;
  float v[OUT_C];
  float m = -3.4e38f;
#pragma unroll
  for (int c = 0; c < OUT_C; ++c) { v[c] = row[c] + b2[c]; m = fmaxf(m, v[c]); }
  float sum = 0.f;
#pragma unroll
  for (int c = 0; c < OUT_C; ++c) sum += expf(v[c] - m);
  float l = logf(sum);
#pragma unroll
  for (int c = 0; c < OUT_C; ++c) row[c] = v[c] - m - l;
}

// ---------------------------------------------------------------------------
extern "C" void kernel_launch(void* const* d_in, const int* in_sizes, int n_in,
                              void* d_out, int out_size, void* d_ws, size_t ws_size,
                              hipStream_t stream) {
  const float*     x   = (const float*)d_in[0];
  const long long* ei  = (const long long*)d_in[1];
  const float*     w1  = (const float*)d_in[2];
  const float*     at_s1 = (const float*)d_in[3];
  const float*     at_d1 = (const float*)d_in[4];
  const float*     b1  = (const float*)d_in[5];
  const float*     w2  = (const float*)d_in[6];
  const float*     at_s2 = (const float*)d_in[7];
  const float*     at_d2 = (const float*)d_in[8];
  const float*     b2  = (const float*)d_in[9];
  float* out = (float*)d_out;

  const int Nn = in_sizes[0] / IN_C;     // 20000
  const int E  = in_sizes[1] / 2;        // 320000
  const int EE = E + Nn;                 // with self loops

  // ---- workspace carve-up (floats) ----
  float* ws = (float*)d_ws;
  float* h1    = ws; ws += (size_t)Nn * HIDC;
  float* as1   = ws; ws += (size_t)Nn * HEADS;
  float* ad1   = ws; ws += (size_t)Nn * HEADS;
  float* eexp1 = ws; ws += (size_t)EE * HEADS;
  float* h2    = ws; ws += (size_t)Nn * OUT_C;
  float* as2   = ws; ws += Nn;
  float* ad2   = ws; ws += Nn;
  float* eexp2 = ws; ws += EE;
  float* zreg  = ws;                      // ---- zero-init region ----
  float* m1    = ws; ws += (size_t)Nn * HEADS;   // max keys (uint)
  float* z1    = ws; ws += (size_t)Nn * HEADS;
  float* out1  = ws; ws += (size_t)Nn * HIDC;    // aggregate, becomes h1e after elu
  float* m2    = ws; ws += Nn;                   // max keys (uint)
  float* z2    = ws; ws += Nn;
  const long long zcount = (long long)(ws - zreg);

  const int TPB = 256;
  const int mtiles = (Nn + 15) / 16;

  // 0) init
  zero_f<<<(int)((zcount + TPB - 1) / TPB), TPB, 0, stream>>>(zreg, zcount);
  zero_f<<<(Nn * OUT_C + TPB - 1) / TPB, TPB, 0, stream>>>(out, (long long)Nn * OUT_C);

  // 1) layer 1
  gemm1_wmma<<<dim3(mtiles), dim3(32, 8), 0, stream>>>(x, w1, h1, Nn);
  node_att1<<<(Nn * HEADS + TPB - 1) / TPB, TPB, 0, stream>>>(h1, at_s1, at_d1, as1, ad1, Nn);
  edge_max1<<<(EE * HEADS + TPB - 1) / TPB, TPB, 0, stream>>>(ei, E, EE, as1, ad1, (unsigned*)m1);
  edge_sum1<<<(EE * HEADS + TPB - 1) / TPB, TPB, 0, stream>>>(ei, E, EE, as1, ad1,
                                                              (const unsigned*)m1, z1, eexp1);
  edge_aggr1<<<(EE * HIDC + TPB - 1) / TPB, TPB, 0, stream>>>(ei, E, EE, eexp1, z1, h1, out1);
  elu_bias<<<(Nn * HIDC + TPB - 1) / TPB, TPB, 0, stream>>>(out1, b1, Nn * HIDC);

  // 2) layer 2
  gemm2_wmma<<<dim3(mtiles, (OUT_C + 15) / 16), dim3(32), 0, stream>>>(out1, w2, h2, Nn);
  node_att2<<<(Nn + TPB - 1) / TPB, TPB, 0, stream>>>(h2, at_s2, at_d2, as2, ad2, Nn);
  edge_max2<<<(EE + TPB - 1) / TPB, TPB, 0, stream>>>(ei, E, EE, as2, ad2, (unsigned*)m2);
  edge_sum2<<<(EE + TPB - 1) / TPB, TPB, 0, stream>>>(ei, E, EE, as2, ad2,
                                                      (const unsigned*)m2, z2, eexp2);
  edge_aggr2<<<(EE * OUT_C + TPB - 1) / TPB, TPB, 0, stream>>>(ei, E, EE, eexp2, z2, h2, out);

  // 3) bias + log_softmax in place
  finalize_lsm<<<(Nn + TPB - 1) / TPB, TPB, 0, stream>>>(out, b2, Nn);
}
